// SnakeRefineModel_20186346291788
// MI455X (gfx1250) — matrix operand order repaired and checked by reading, not compiled, not run on hardware
//
#include <hip/hip_runtime.h>
#include <cstddef>
#include <cstdint>

typedef _Float16 v16h __attribute__((ext_vector_type(16)));
typedef _Float16 v8h  __attribute__((ext_vector_type(8)));
typedef float    v8f  __attribute__((ext_vector_type(8)));

static constexpr int kB  = 8;
static constexpr int kC0 = 192;
static constexpr int kH  = 128, kW = 128, kHW = kH * kW;
static constexpr int kN  = 128;   // points per polygon

// ---------------------------------------------------------------------------
// gfx1250 async global->LDS staging (ASYNCcnt path), with safe fallback.
// Builtin signature (from compiler diagnostic): first parameter is
// non-const v4i* in the global (__device__/addrspace(1)) address space.
// ---------------------------------------------------------------------------
#if defined(__AMDGCN__) && __has_builtin(__builtin_amdgcn_global_load_async_to_lds_b128)
#define HAVE_ASYNC 1
#else
#define HAVE_ASYNC 0
#endif

#if HAVE_ASYNC
typedef int v4i __attribute__((vector_size(4 * sizeof(int))));
typedef __attribute__((address_space(1))) v4i* gv4i_p;
typedef __attribute__((address_space(3))) v4i* lv4i_p;

__device__ inline void async_copy_b128(const _Float16* g, _Float16* l) {
  __builtin_amdgcn_global_load_async_to_lds_b128(
      (gv4i_p)(_Float16*)g, (lv4i_p)l, 0, 0);
}
__device__ inline void wait_async0() {
#if __has_builtin(__builtin_amdgcn_s_wait_asynccnt)
  __builtin_amdgcn_s_wait_asynccnt(0);
#else
  asm volatile("s_wait_asynccnt 0x0" ::: "memory");
#endif
}
#endif

// ---------------------------------------------------------------------------
// WMMA helpers (wave32, 16x16x32 f16 -> f32)
// ---------------------------------------------------------------------------
__device__ inline v16h mk16(v8h lo, v8h hi) {
  return __builtin_shufflevector(lo, hi, 0,1,2,3,4,5,6,7,8,9,10,11,12,13,14,15);
}
__device__ inline v8f wmma_f16(v16h a, v16h b, v8f c) {
  return __builtin_amdgcn_wmma_f32_16x16x32_f16(false, a, false, b, (short)0, c,
                                                false, false);
}

// ---------------------------------------------------------------------------
// Weight conversion f32 -> f16 with K padding (dst [M][Kp], src [M][Ks])
// ---------------------------------------------------------------------------
__global__ void __launch_bounds__(256) cvt_pad(const float* __restrict__ src,
                                               _Float16* __restrict__ dst,
                                               int Ks, int Kp, int total) {
  for (int i = blockIdx.x * blockDim.x + threadIdx.x; i < total;
       i += gridDim.x * blockDim.x) {
    int m = i / Kp, k = i - m * Kp;
    dst[i] = (k < Ks) ? (_Float16)src[(size_t)m * Ks + k] : (_Float16)0.f;
  }
}

__global__ void __launch_bounds__(256) copyf(const float* __restrict__ s,
                                             float* __restrict__ d, int n) {
  int i = blockIdx.x * blockDim.x + threadIdx.x;
  if (i < n) d[i] = s[i];
}

// ---------------------------------------------------------------------------
// 3x3 conv 192->256 + bias + ReLU, implicit GEMM (K = 1728).
// Output y1[b][p][256] f16 (K-contiguous for the following 1x1 GEMM).
// Grid: (HW/16, 256/64, B), 128 threads (4 waves, each one 16x16 WMMA tile).
// Weights staged per-wave into LDS via async global->LDS; im2col tile built
// cooperatively in LDS.
// ---------------------------------------------------------------------------
__global__ void __launch_bounds__(128) conv3x3_head(
    const float* __restrict__ x, const _Float16* __restrict__ Wf,
    const float* __restrict__ bias, _Float16* __restrict__ y1) {
  const int b    = blockIdx.z;
  const int wave = threadIdx.x >> 5;
  const int lane = threadIdx.x & 31;
  const int m0   = blockIdx.y * 64 + wave * 16;
  const int p0   = blockIdx.x * 16;

  __shared__ __align__(16) _Float16 Bt[16][40];
#if HAVE_ASYNC
  __shared__ __align__(16) _Float16 Asr[4][16][32];
  const int srow = lane & 15;
  const int schk = lane >> 4;  // 16B chunk 0/1 (+2 for second op)
#else
  const int alo = (lane & 16) ? 8 : 0;
  const _Float16* ap = Wf + (size_t)(m0 + (lane & 15)) * 1728 + alo;
#endif
  const int bc  = lane & 15;
  const int blo = (lane & 16) ? 8 : 0;

  // gather-lane decomposition: K-lane fixed per thread across the K loop
  const int kloc = threadIdx.x & 31;
  const int nb   = threadIdx.x >> 5;  // columns nb, nb+4, nb+8, nb+12

  v8f acc = {};
  for (int kk = 0; kk < 1728; kk += 32) {
#if HAVE_ASYNC
    {  // stage this wave's 16x32 A tile (two async b128 ops, 512B each)
      const _Float16* g0 =
          Wf + (size_t)(m0 + srow) * 1728 + kk + schk * 8;
      _Float16* l0 = &Asr[wave][srow][schk * 8];
      async_copy_b128(g0, l0);
      async_copy_b128(g0 + 16, l0 + 16);
    }
#endif
    // cooperative im2col gather: 16 cols x 32 K elements
    {
      int kg  = kk + kloc;
      int ci  = kg / 9;
      int rem = kg - ci * 9;
      int kh  = rem / 3 - 1;
      int kw  = rem - (rem / 3) * 3 - 1;
      const float* xc = x + (((size_t)b * kC0 + ci) << 14);
#pragma unroll
      for (int u = 0; u < 4; ++u) {
        int n  = nb + 4 * u;
        int p  = p0 + n;
        int hh = (p >> 7) + kh;
        int ww = (p & 127) + kw;
        float v = 0.f;
        if (hh >= 0 && hh < kH && ww >= 0 && ww < kW)
          v = xc[(hh << 7) + ww];
        Bt[n][kloc] = (_Float16)v;
      }
    }
    __syncthreads();
#if HAVE_ASYNC
    wait_async0();
    const _Float16* apl = &Asr[wave][lane & 15][(lane & 16) ? 8 : 0];
    v8h a0 = *(const v8h*)apl;
    v8h a1 = *(const v8h*)(apl + 16);
#else
    v8h a0 = *(const v8h*)(ap + kk);
    v8h a1 = *(const v8h*)(ap + kk + 16);
#endif
    const _Float16* lp = &Bt[bc][blo];
    v8h b0 = *(const v8h*)lp;
    v8h b1 = *(const v8h*)(lp + 16);
    acc = wmma_f16(mk16(a0, a1), mk16(b0, b1), acc);
    __syncthreads();
  }
  const int col = p0 + (lane & 15);
  const int ch  = m0 + ((lane & 16) ? 8 : 0);
  v8h o;
  for (int r = 0; r < 8; ++r) {
    float v = acc[r] + bias[ch + r];
    o[r] = (_Float16)(v > 0.f ? v : 0.f);
  }
  *(v8h*)(y1 + ((size_t)b * kHW + col) * 256 + ch) = o;
}

// ---------------------------------------------------------------------------
// Generic GEMM: A[M][K] f16 x Bt[b][NB][K] f16 -> out[b][col][ostride]+ch.
// K % 32 == 0. No LDS: per-lane fragments straight from global.
// Grid: (NB/16, M/64, B), 128 threads.
// ---------------------------------------------------------------------------
template <bool RELU, bool OUTF32>
__global__ void __launch_bounds__(128) gemm_bt(
    const _Float16* __restrict__ A, const _Float16* __restrict__ Bt,
    const float* __restrict__ bias, int K, int NB, float* __restrict__ outF,
    _Float16* __restrict__ outH, int ostride, int ochoff) {
  const int b    = blockIdx.z;
  const int wave = threadIdx.x >> 5;
  const int lane = threadIdx.x & 31;
  const int m0   = blockIdx.y * 64 + wave * 16;
  const int n0   = blockIdx.x * 16;

  const _Float16* Bb = Bt + (size_t)b * NB * K;
  const int alo = (lane & 16) ? 8 : 0;
  const int blo = (lane & 16) ? 8 : 0;
  const _Float16* ap = A + (size_t)(m0 + (lane & 15)) * K + alo;
  const _Float16* bp = Bb + (size_t)(n0 + (lane & 15)) * K + blo;

  v8f acc = {};
  for (int kk = 0; kk < K; kk += 32) {
    if (kk + 64 <= K) {  // gfx1250 global_prefetch_b8
      __builtin_prefetch(ap + kk + 32, 0, 0);
      __builtin_prefetch(bp + kk + 32, 0, 0);
    }
    v8h a0 = *(const v8h*)(ap + kk);
    v8h a1 = *(const v8h*)(ap + kk + 16);
    v8h b0 = *(const v8h*)(bp + kk);
    v8h b1 = *(const v8h*)(bp + kk + 16);
    acc = wmma_f16(mk16(a0, a1), mk16(b0, b1), acc);
  }
  const int col = n0 + (lane & 15);
  const int ch  = m0 + ((lane & 16) ? 8 : 0);
  size_t obase  = ((size_t)b * NB + col) * ostride + ochoff + ch;
  if constexpr (OUTF32) {
    v8f o;
    for (int r = 0; r < 8; ++r) {
      float v = acc[r] + (bias ? bias[ch + r] : 0.f);
      if (RELU) v = fmaxf(v, 0.f);
      o[r] = v;
    }
    *(v8f*)(outF + obase) = o;
  } else {
    v8h o;
    for (int r = 0; r < 8; ++r) {
      float v = acc[r] + (bias ? bias[ch + r] : 0.f);
      if (RELU) v = fmaxf(v, 0.f);
      o[r] = (_Float16)v;
    }
    *(v8h*)(outH + obase) = o;
  }
}

// ---------------------------------------------------------------------------
// Circular dilated conv1d block: GEMM (256 x 128, K = cin*9) with circular
// im2col gather into LDS; post-op = bias -> ReLU -> gamma*x+beta (+residual).
// act layout [b][n][actW] f16; output states[b][n][2048] at choff_out.
// Grid: (128/16, 256/64, B), 128 threads.
// ---------------------------------------------------------------------------
__global__ void __launch_bounds__(128) snake_block(
    const _Float16* __restrict__ act, int actW, int choff_in, int cin, int dil,
    const _Float16* __restrict__ Wf, const float* __restrict__ bias,
    const float* __restrict__ gamma, const float* __restrict__ beta,
    int do_res, _Float16* __restrict__ states, int choff_out) {
  const int b    = blockIdx.z;
  const int wave = threadIdx.x >> 5;
  const int lane = threadIdx.x & 31;
  const int m0   = blockIdx.y * 64 + wave * 16;
  const int n0   = blockIdx.x * 16;
  const int K    = cin * 9;
  const int Kpad = (K + 31) & ~31;

  __shared__ __align__(16) _Float16 Bt[16][40];
#if HAVE_ASYNC
  __shared__ __align__(16) _Float16 Asr[4][16][32];
  const int srow = lane & 15;
  const int schk = lane >> 4;
#else
  const int alo = (lane & 16) ? 8 : 0;
  const _Float16* ap = Wf + (size_t)(m0 + (lane & 15)) * Kpad + alo;
#endif
  const int bc  = lane & 15;
  const int blo = (lane & 16) ? 8 : 0;

  const int kloc = threadIdx.x & 31;
  const int nb   = threadIdx.x >> 5;

  v8f acc = {};
  for (int kk = 0; kk < Kpad; kk += 32) {
#if HAVE_ASYNC
    {
      const _Float16* g0 =
          Wf + (size_t)(m0 + srow) * Kpad + kk + schk * 8;
      _Float16* l0 = &Asr[wave][srow][schk * 8];
      async_copy_b128(g0, l0);
      async_copy_b128(g0 + 16, l0 + 16);
    }
#endif
    {
      int kg = kk + kloc;
      int ci = kg / 9;
      int j  = kg - ci * 9;
      int joff = (j - 4) * dil;
      const _Float16* ac = act + (size_t)b * kN * actW + choff_in + ci;
      bool inK = (kg < K);
#pragma unroll
      for (int u = 0; u < 4; ++u) {
        int n  = nb + 4 * u;
        int nn = (n0 + n + joff) & 127;  // circular index
        _Float16 v = (_Float16)0.f;
        if (inK) v = ac[(size_t)nn * actW];
        Bt[n][kloc] = v;
      }
    }
    __syncthreads();
#if HAVE_ASYNC
    wait_async0();
    const _Float16* apl = &Asr[wave][lane & 15][(lane & 16) ? 8 : 0];
    v8h a0 = *(const v8h*)apl;
    v8h a1 = *(const v8h*)(apl + 16);
#else
    v8h a0 = *(const v8h*)(ap + kk);
    v8h a1 = *(const v8h*)(ap + kk + 16);
#endif
    const _Float16* lp = &Bt[bc][blo];
    v8h b0 = *(const v8h*)lp;
    v8h b1 = *(const v8h*)(lp + 16);
    acc = wmma_f16(mk16(a0, a1), mk16(b0, b1), acc);
    __syncthreads();
  }
  const int col = n0 + (lane & 15);
  const int ch  = m0 + ((lane & 16) ? 8 : 0);
  const _Float16* rp = act + ((size_t)b * kN + col) * actW + choff_in + ch;
  v8h o;
  for (int r = 0; r < 8; ++r) {
    float v = acc[r] + bias[ch + r];
    v = v > 0.f ? v : 0.f;
    v = v * gamma[ch + r] + beta[ch + r];
    if (do_res) v += (float)rp[r];
    o[r] = (_Float16)v;
  }
  *(v8h*)(states + ((size_t)b * kN + col) * 2048 + choff_out + ch) = o;
}

// ---------------------------------------------------------------------------
// Per-batch min over valid points (for img_poly_to_can_poly).
// ---------------------------------------------------------------------------
__global__ void __launch_bounds__(128) mins_kernel(const float* __restrict__ py,
                                                   const float* __restrict__ mask,
                                                   float* __restrict__ mins) {
  int b = blockIdx.x, n = threadIdx.x;
  float m  = mask[b * kN + n];
  float vx = m > 0.f ? py[(b * kN + n) * 2]     : 1e8f;
  float vy = m > 0.f ? py[(b * kN + n) * 2 + 1] : 1e8f;
  __shared__ float sx[128], sy[128];
  sx[n] = vx; sy[n] = vy;
  __syncthreads();
  for (int s = 64; s > 0; s >>= 1) {
    if (n < s) { sx[n] = fminf(sx[n], sx[n + s]); sy[n] = fminf(sy[n], sy[n + s]); }
    __syncthreads();
  }
  if (n == 0) { mins[b * 2] = sx[0]; mins[b * 2 + 1] = sy[0]; }
}

// Bilinear sample (align_corners=True, zero padding), pfeat [b][hw][64] f32.
__device__ inline void bilinear64(const float* __restrict__ fb, float px,
                                  float pyv, float* out) {
  float xx  = px  * (127.f / 128.f);
  float yy  = pyv * (127.f / 128.f);
  float x0f = floorf(xx), y0f = floorf(yy);
  float wx  = xx - x0f,   wy  = yy - y0f;
  int x0 = (int)x0f, y0 = (int)y0f, x1 = x0 + 1, y1 = y0 + 1;
  float m00 = (x0 >= 0 && x0 < kW && y0 >= 0 && y0 < kH) ? 1.f : 0.f;
  float m10 = (x1 >= 0 && x1 < kW && y0 >= 0 && y0 < kH) ? 1.f : 0.f;
  float m01 = (x0 >= 0 && x0 < kW && y1 >= 0 && y1 < kH) ? 1.f : 0.f;
  float m11 = (x1 >= 0 && x1 < kW && y1 >= 0 && y1 < kH) ? 1.f : 0.f;
  int cx0 = x0 < 0 ? 0 : (x0 > 127 ? 127 : x0);
  int cx1 = x1 < 0 ? 0 : (x1 > 127 ? 127 : x1);
  int cy0 = y0 < 0 ? 0 : (y0 > 127 ? 127 : y0);
  int cy1 = y1 < 0 ? 0 : (y1 > 127 ? 127 : y1);
  const float* p00 = fb + ((size_t)(cy0 * kW + cx0)) * 64;
  const float* p10 = fb + ((size_t)(cy0 * kW + cx1)) * 64;
  const float* p01 = fb + ((size_t)(cy1 * kW + cx0)) * 64;
  const float* p11 = fb + ((size_t)(cy1 * kW + cx1)) * 64;
  float w00 = (1.f - wx) * (1.f - wy) * m00, w10 = wx * (1.f - wy) * m10;
  float w01 = (1.f - wx) * wy * m01,         w11 = wx * wy * m11;
#pragma unroll 8
  for (int cc = 0; cc < 64; ++cc)
    out[cc] = p00[cc] * w00 + p10[cc] * w10 + p01[cc] * w01 + p11[cc] * w11;
}

// Build snake input xin[b][n][66] = [feat*mask (64), c_py (2)] f16.
__global__ void __launch_bounds__(128) sample_xin(
    const float* __restrict__ py, const float* __restrict__ mask,
    const float* __restrict__ mins, const float* __restrict__ pfeat,
    _Float16* __restrict__ xin) {
  int b = blockIdx.x, n = threadIdx.x;
  float px = py[(b * kN + n) * 2], pyv = py[(b * kN + n) * 2 + 1];
  float m  = mask[b * kN + n];
  float feat[64];
  bilinear64(pfeat + (size_t)b * kHW * 64, px, pyv, feat);
  _Float16* xr = xin + ((size_t)b * kN + n) * 66;
  for (int cc = 0; cc < 64; ++cc) xr[cc] = (_Float16)(feat[cc] * m);
  xr[64] = (_Float16)((px  - mins[b * 2])     * m);
  xr[65] = (_Float16)((pyv - mins[b * 2 + 1]) * m);
}

// glob = max_n fusion(state); cat[b][n][2304] = [glob, states].
__global__ void __launch_bounds__(256) fusion_max_cat(
    const float* __restrict__ fus, const _Float16* __restrict__ states,
    _Float16* __restrict__ cat) {
  int b = blockIdx.x, cth = threadIdx.x;
  __shared__ float g[256];
  float mx = -3.0e38f;
  for (int n = 0; n < kN; ++n)
    mx = fmaxf(mx, fus[((size_t)b * kN + n) * 256 + cth]);
  g[cth] = mx;
  __syncthreads();
  for (int t = cth; t < kN * 2304; t += 256) {
    int n = t / 2304;
    int k = t - n * 2304;
    _Float16 v = (k < 256) ? (_Float16)g[k]
                           : states[((size_t)b * kN + n) * 2048 + (k - 256)];
    cat[((size_t)b * kN + n) * 2304 + k] = v;
  }
}

// Final 64->2 offset head + py update (py += off*mask*STRIDE).
__global__ void __launch_bounds__(128) pred3_update(
    const _Float16* __restrict__ p2o, const float* __restrict__ w3,
    const float* __restrict__ b3, const float* __restrict__ mask,
    float* __restrict__ py) {
  int b = blockIdx.x, n = threadIdx.x;
  float o0 = b3[0], o1 = b3[1];
  const _Float16* r = p2o + ((size_t)b * kN + n) * 64;
  for (int cc = 0; cc < 64; ++cc) {
    float f = (float)r[cc];
    o0 += w3[cc] * f;
    o1 += w3[64 + cc] * f;
  }
  float m = mask[b * kN + n] * 4.0f;  // STRIDE
  py[(b * kN + n) * 2]     += o0 * m;
  py[(b * kN + n) * 2 + 1] += o1 * m;
}

// Classification sampling: pf[b][c*128+n] = sample(cls_feat)*mask.
__global__ void __launch_bounds__(128) sample_pf(
    const float* __restrict__ py, const float* __restrict__ mask,
    const float* __restrict__ pfeat, float* __restrict__ pf) {
  int b = blockIdx.x, n = threadIdx.x;
  float feat[64];
  bilinear64(pfeat + (size_t)b * kHW * 64, py[(b * kN + n) * 2],
             py[(b * kN + n) * 2 + 1], feat);
  float m = mask[b * kN + n];
  for (int cc = 0; cc < 64; ++cc)
    pf[(size_t)b * 8192 + cc * kN + n] = feat[cc] * m;
}

// cls = pf @ Wp.T + bp    (one block per (j, b), K = 8192)
__global__ void __launch_bounds__(128) cls1_kernel(
    const float* __restrict__ pf, const float* __restrict__ Wp,
    const float* __restrict__ bp, float* __restrict__ tmp) {
  int j = blockIdx.x, b = blockIdx.y, t = threadIdx.x;
  const float* pr = pf + (size_t)b * 8192;
  const float* wr = Wp + (size_t)j * 8192;
  float s = 0.f;
  for (int k = t; k < 8192; k += 128) s += pr[k] * wr[k];
  __shared__ float red[128];
  red[t] = s;
  __syncthreads();
  for (int ss = 64; ss > 0; ss >>= 1) {
    if (t < ss) red[t] += red[t + ss];
    __syncthreads();
  }
  if (t == 0) tmp[b * kN + j] = red[0] + bp[j];
}

// cls2 = (cls @ Wf.T + bf) * mask   -> d_out tail
__global__ void __launch_bounds__(128) cls2_kernel(
    const float* __restrict__ tmp, const float* __restrict__ Wfz,
    const float* __restrict__ bf, const float* __restrict__ mask,
    float* __restrict__ outc) {
  int b = blockIdx.x, j = threadIdx.x;
  float s = bf[j];
  for (int i = 0; i < kN; ++i) s += tmp[b * kN + i] * Wfz[(size_t)j * kN + i];
  outc[b * kN + j] = s * mask[b * kN + j];
}

__global__ void __launch_bounds__(128) write_py(const float* __restrict__ py,
                                                float* __restrict__ out) {
  int i = blockIdx.x * blockDim.x + threadIdx.x;
  if (i < kB * kN * 2) out[i] = py[i] * 4.0f;  // DOWN_RATIO
}

// ---------------------------------------------------------------------------
// Host orchestration
// ---------------------------------------------------------------------------
extern "C" void kernel_launch(void* const* d_in, const int* in_sizes, int n_in,
                              void* d_out, int out_size, void* d_ws,
                              size_t ws_size, hipStream_t stream) {
  (void)in_sizes; (void)n_in; (void)out_size; (void)ws_size;

  auto F = [&](int i) { return (const float*)d_in[i]; };
  // top-level setup_inputs() dict order:
  const float* x_feat   = F(0);  // transformer_features (B,192,128,128)
  const float* pred_pts = F(1);  // pred_points (B,128,2)
  const float* vmask    = F(2);  // valid_mask (B,128)

  // params tree-flattened (alphabetical dict keys, list order):
  int c = 3;
  const float* cls_fuse_b = F(c++);
  const float* cls_fuse_w = F(c++);
  const float* cls_poly_b = F(c++);
  const float* cls_poly_w = F(c++);
  struct Snk {
    const float *fus_b, *fus_w, *h_b, *h_beta, *h_gamma, *h_w;
    const float *p_b1, *p_b2, *p_b3, *p_w1, *p_w2, *p_w3;
    const float *r_b[7], *r_beta[7], *r_gamma[7], *r_w[7];
  } S[3];
  for (int s = 0; s < 3; ++s) {
    S[s].fus_b = F(c++); S[s].fus_w = F(c++);
    S[s].h_b = F(c++); S[s].h_beta = F(c++); S[s].h_gamma = F(c++); S[s].h_w = F(c++);
    S[s].p_b1 = F(c++); S[s].p_b2 = F(c++); S[s].p_b3 = F(c++);
    S[s].p_w1 = F(c++); S[s].p_w2 = F(c++); S[s].p_w3 = F(c++);
    for (int j = 0; j < 7; ++j) {
      S[s].r_b[j] = F(c++); S[s].r_beta[j] = F(c++);
      S[s].r_gamma[j] = F(c++); S[s].r_w[j] = F(c++);
    }
  }
  const float *tf_b1 = F(c++), *tf_b2 = F(c++), *tf_w1 = F(c++), *tf_w2 = F(c++);
  const float *tfc_b1 = F(c++), *tfc_b2 = F(c++), *tfc_w1 = F(c++), *tfc_w2 = F(c++);

  // workspace bump allocator (256B aligned)
  char* base = (char*)d_ws;
  size_t off = 0;
  auto alloc = [&](size_t bytes) -> char* {
    size_t a = (off + 255) & ~(size_t)255;
    off = a + bytes;
    return base + a;
  };
  _Float16* hw1   = (_Float16*)alloc((size_t)256 * 1728 * 2);
  _Float16* hw2   = (_Float16*)alloc((size_t)64 * 256 * 2);
  _Float16* y1    = (_Float16*)alloc((size_t)kB * kHW * 256 * 2);
  float*    pfeat = (float*)   alloc((size_t)kB * kHW * 64 * 4);
  _Float16 *sw_head[3], *sw_res[3][7], *sw_fus[3], *sw_p1[3], *sw_p2[3];
  for (int s = 0; s < 3; ++s) {
    sw_head[s] = (_Float16*)alloc((size_t)256 * 608 * 2);
    for (int j = 0; j < 7; ++j)
      sw_res[s][j] = (_Float16*)alloc((size_t)256 * 2304 * 2);
    sw_fus[s] = (_Float16*)alloc((size_t)256 * 2048 * 2);
    sw_p1[s]  = (_Float16*)alloc((size_t)256 * 2304 * 2);
    sw_p2[s]  = (_Float16*)alloc((size_t)64 * 256 * 2);
  }
  float*    pyw    = (float*)   alloc((size_t)kB * kN * 2 * 4);
  float*    minsb  = (float*)   alloc((size_t)kB * 2 * 4);
  _Float16* xin    = (_Float16*)alloc((size_t)kB * kN * 66 * 2);
  _Float16* states = (_Float16*)alloc((size_t)kB * kN * 2048 * 2);
  float*    fusb   = (float*)   alloc((size_t)kB * kN * 256 * 4);
  _Float16* catb   = (_Float16*)alloc((size_t)kB * kN * 2304 * 2);
  _Float16* p1o    = (_Float16*)alloc((size_t)kB * kN * 256 * 2);
  _Float16* p2o    = (_Float16*)alloc((size_t)kB * kN * 64 * 2);
  float*    pfb    = (float*)   alloc((size_t)kB * 8192 * 4);
  float*    clst   = (float*)   alloc((size_t)kB * kN * 4);

  auto cvt = [&](const float* src, _Float16* dst, int M, int Ks, int Kp) {
    int total = M * Kp;
    cvt_pad<<<(total + 255) / 256, 256, 0, stream>>>(src, dst, Ks, Kp, total);
  };

  // Convert all weights to f16 (snake head padded 594 -> 608).
  cvt(tf_w1, hw1, 256, 1728, 1728);
  cvt(tf_w2, hw2, 64, 256, 256);
  for (int s = 0; s < 3; ++s) {
    cvt(S[s].h_w, sw_head[s], 256, 594, 608);
    for (int j = 0; j < 7; ++j) cvt(S[s].r_w[j], sw_res[s][j], 256, 2304, 2304);
    cvt(S[s].fus_w, sw_fus[s], 256, 2048, 2048);
    cvt(S[s].p_w1, sw_p1[s], 256, 2304, 2304);
    cvt(S[s].p_w2, sw_p2[s], 64, 256, 256);
  }
  copyf<<<(kB * kN * 2 + 255) / 256, 256, 0, stream>>>(pred_pts, pyw, kB * kN * 2);

  // trans head (tf): conv3x3+ReLU then 1x1 -> poly_feat [b][hw][64]
  conv3x3_head<<<dim3(kHW / 16, 4, kB), 128, 0, stream>>>(x_feat, hw1, tf_b1, y1);
  gemm_bt<false, true><<<dim3(kHW / 16, 1, kB), 128, 0, stream>>>(
      hw2, y1, tf_b2, 256, kHW, pfeat, nullptr, 64, 0);

  static const int DIL[7] = {1, 1, 1, 2, 2, 4, 4};
  for (int it = 0; it < 3; ++it) {
    mins_kernel<<<kB, 128, 0, stream>>>(pyw, vmask, minsb);
    sample_xin<<<kB, 128, 0, stream>>>(pyw, vmask, minsb, pfeat, xin);
    // head block (66 -> 256, dil 1) -> states seg 0
    snake_block<<<dim3(8, 4, kB), 128, 0, stream>>>(
        xin, 66, 0, 66, 1, sw_head[it], S[it].h_b, S[it].h_gamma, S[it].h_beta,
        0, states, 0);
    // 7 residual blocks
    for (int j = 0; j < 7; ++j)
      snake_block<<<dim3(8, 4, kB), 128, 0, stream>>>(
          states, 2048, j * 256, 256, DIL[j], sw_res[it][j], S[it].r_b[j],
          S[it].r_gamma[j], S[it].r_beta[j], 1, states, (j + 1) * 256);
    // fusion 2048 -> 256, then max over points + concat
    gemm_bt<false, true><<<dim3(8, 4, kB), 128, 0, stream>>>(
        sw_fus[it], states, S[it].fus_b, 2048, kN, fusb, nullptr, 256, 0);
    fusion_max_cat<<<kB, 256, 0, stream>>>(fusb, states, catb);
    // prediction MLP
    gemm_bt<true, false><<<dim3(8, 4, kB), 128, 0, stream>>>(
        sw_p1[it], catb, S[it].p_b1, 2304, kN, nullptr, p1o, 256, 0);
    gemm_bt<true, false><<<dim3(8, 1, kB), 128, 0, stream>>>(
        sw_p2[it], p1o, S[it].p_b2, 256, kN, nullptr, p2o, 64, 0);
    pred3_update<<<kB, 128, 0, stream>>>(p2o, S[it].p_w3, S[it].p_b3, vmask, pyw);
  }

  // classification head (tfc) reuses head buffers
  cvt(tfc_w1, hw1, 256, 1728, 1728);
  cvt(tfc_w2, hw2, 64, 256, 256);
  conv3x3_head<<<dim3(kHW / 16, 4, kB), 128, 0, stream>>>(x_feat, hw1, tfc_b1, y1);
  gemm_bt<false, true><<<dim3(kHW / 16, 1, kB), 128, 0, stream>>>(
      hw2, y1, tfc_b2, 256, kHW, pfeat, nullptr, 64, 0);

  sample_pf<<<kB, 128, 0, stream>>>(pyw, vmask, pfeat, pfb);
  cls1_kernel<<<dim3(128, kB), 128, 0, stream>>>(pfb, cls_poly_w, cls_poly_b, clst);
  cls2_kernel<<<kB, 128, 0, stream>>>(clst, cls_fuse_w, cls_fuse_b, vmask,
                                      (float*)d_out + kB * kN * 2);
  write_py<<<(kB * kN * 2 + 127) / 128, 128, 0, stream>>>(pyw, (float*)d_out);
}
